// Model_35098472743206
// MI455X (gfx1250) — matrix operand order repaired and checked by reading, not compiled, not run on hardware
//
#include <hip/hip_runtime.h>
#include <hip/hip_bf16.h>
#include <math.h>

// ---------------- model dims ----------------
#define BB   4
#define SS   2048
#define MM   (BB * SS)       // 8192 rows
#define DD   384
#define QKD  48
#define QKP  64              // qh/kh padded K
#define EE   768
#define NEXP 1632
#define NEXP_P 1664          // expand N padded to 13*128
#define NVOC 100
#define NVOC_P 128

typedef _Float16 half16 __attribute__((ext_vector_type(16)));
typedef float    f32x8  __attribute__((ext_vector_type(8)));

static __device__ inline f32x8 wmma_f16(half16 a, half16 b, f32x8 c) {
    // v_wmma_f32_16x16x32_f16  (8-arg form: neg_a, A, neg_b, B, c_mod, C, reuse_a, reuse_b)
    return __builtin_amdgcn_wmma_f32_16x16x32_f16(false, a, false, b, (short)0, c, false, false);
}

static __device__ inline f32x8 zero8() {
    f32x8 z;
    for (int i = 0; i < 8; ++i) z[i] = 0.0f;
    return z;
}

// A fragment: 16(M) x 32(K) f16, row-major source with leading dim lda (elements).
// lane L<16  : row L, K 0..7   (vgpr0-3) and K 16..23 (vgpr4-7)
// lane L+16  : row L, K 8..15             and K 24..31
static __device__ inline half16 load_frag_a(const _Float16* base, int lda) {
    int lane = threadIdx.x & 31;
    int row  = lane & 15;
    int koff = (lane < 16) ? 0 : 8;
    const _Float16* p = base + (size_t)row * lda + koff;
    half16 r;
    __builtin_memcpy(&r, p, 16);                         // 8 halves
    __builtin_memcpy((char*)&r + 16, p + 16, 16);        // 8 halves, K+16
    return r;
}

// B fragment: 32(K) x 16(N) f16 where the source is W stored (N,K) row-major:
// column n of B == row n of W (contiguous in K).
// lane n<16 : W[n][k0..k0+7],W[n][k0+16..k0+23]   lane n+16 : W[n][k0+8..k0+15],W[n][k0+24..k0+31]
static __device__ inline half16 load_frag_b(const _Float16* base, int ldb) {
    int lane = threadIdx.x & 31;
    int n    = lane & 15;
    int koff = (lane < 16) ? 0 : 8;
    const _Float16* p = base + (size_t)n * ldb + koff;
    half16 r;
    __builtin_memcpy(&r, p, 16);
    __builtin_memcpy((char*)&r + 16, p + 16, 16);
    return r;
}

// C fragment layout: vgpr r, lanes 0-15 -> row r, col lane; lanes 16-31 -> row r+8.
static __device__ inline void store_frag(float* C, int ldc, int m0, int n0,
                                         f32x8 acc, int Nstore, int residual) {
    int lane = threadIdx.x & 31;
    int hf   = lane >> 4;
    int col  = lane & 15;
    if (n0 + col < Nstore) {
        for (int r = 0; r < 8; ++r) {
            size_t idx = (size_t)(m0 + r + hf * 8) * ldc + n0 + col;
            if (residual) C[idx] += acc[r]; else C[idx] = acc[r];
        }
    }
}

// ---------------- weight conversion / padding ----------------
__global__ void conv_expw_kernel(const float* __restrict__ src, _Float16* __restrict__ dst) {
    size_t idx = (size_t)blockIdx.x * 256 + threadIdx.x;
    size_t total = (size_t)8 * NEXP_P * DD;
    if (idx >= total) return;
    int c  = idx % DD;
    size_t r = idx / DD;
    int nb = r / NEXP_P, rr = r % NEXP_P;
    float v = (rr < NEXP) ? src[((size_t)nb * NEXP + rr) * DD + c] : 0.0f;
    dst[idx] = (_Float16)v;
}

__global__ void conv_projw_kernel(const float* __restrict__ src, _Float16* __restrict__ dst) {
    size_t idx = (size_t)blockIdx.x * 256 + threadIdx.x;
    if (idx >= (size_t)8 * DD * EE) return;
    dst[idx] = (_Float16)src[idx];
}

__global__ void conv_outw_kernel(const float* __restrict__ src, _Float16* __restrict__ dst) {
    size_t idx = (size_t)blockIdx.x * 256 + threadIdx.x;
    if (idx >= (size_t)NVOC_P * DD) return;
    int r = idx / DD, c = idx % DD;
    dst[idx] = (_Float16)((r < NVOC) ? src[(size_t)r * DD + c] : 0.0f);
}

// ---------------- embedding + LN -> x (f32) ----------------
__global__ __launch_bounds__(128) void embed_ln_kernel(const int* __restrict__ q,
                                                       const float* __restrict__ freqs,
                                                       const float* __restrict__ g,
                                                       const float* __restrict__ b,
                                                       float* __restrict__ x) {
    int row = blockIdx.x;
    int tid = threadIdx.x;
    float tok = (float)q[row];
    float v[3], s = 0.f, s2 = 0.f;
    for (int t = 0; t < 3; ++t) {
        int i = tid + t * 128;
        int j = (i < DD / 2) ? i : i - DD / 2;
        float qe = tok * freqs[j];
        float vv = (i < DD / 2) ? sinf(qe) : cosf(qe);
        v[t] = vv; s += vv; s2 += vv * vv;
    }
    __shared__ float rs[128], rq[128];
    rs[tid] = s; rq[tid] = s2; __syncthreads();
    for (int o = 64; o > 0; o >>= 1) {
        if (tid < o) { rs[tid] += rs[tid + o]; rq[tid] += rq[tid + o]; }
        __syncthreads();
    }
    float mean = rs[0] / DD;
    float var  = rq[0] / DD - mean * mean;
    float inv  = rsqrtf(var + 1e-5f);
    for (int t = 0; t < 3; ++t) {
        int i = tid + t * 128;
        x[(size_t)row * DD + i] = (v[t] - mean) * inv * g[i] + b[i];
    }
}

// ---------------- LN (f32 -> f16) ----------------
__global__ __launch_bounds__(128) void ln_f16_kernel(const float* __restrict__ x,
                                                     const float* __restrict__ g,
                                                     const float* __restrict__ b,
                                                     _Float16* __restrict__ y) {
    int row = blockIdx.x;
    int tid = threadIdx.x;
    const float* xr = x + (size_t)row * DD;
    float v[3], s = 0.f, s2 = 0.f;
    for (int t = 0; t < 3; ++t) {
        float vv = xr[tid + t * 128];
        v[t] = vv; s += vv; s2 += vv * vv;
    }
    __shared__ float rs[128], rq[128];
    rs[tid] = s; rq[tid] = s2; __syncthreads();
    for (int o = 64; o > 0; o >>= 1) {
        if (tid < o) { rs[tid] += rs[tid + o]; rq[tid] += rq[tid + o]; }
        __syncthreads();
    }
    float mean = rs[0] / DD;
    float var  = rq[0] / DD - mean * mean;
    float inv  = rsqrtf(var + 1e-5f);
    for (int t = 0; t < 3; ++t) {
        int i = tid + t * 128;
        y[(size_t)row * DD + i] = (_Float16)((v[t] - mean) * inv * g[i] + b[i]);
    }
}

// ---------------- generic WMMA GEMM: C[M,N](f32) = A[M,K](f16) @ W[N,K]^T(f16) ----------------
// block = 256 threads = 8 waves, waves arranged 2(M) x 4(N), wave tile 32x32.
__global__ __launch_bounds__(256) void gemm_kernel(const _Float16* __restrict__ A,
                                                   const _Float16* __restrict__ W,
                                                   float* __restrict__ C,
                                                   int K, int ldc, int Nstore, int residual) {
    int wave = threadIdx.x >> 5;
    int wm = wave >> 2, wn = wave & 3;
    int m0 = blockIdx.y * 64 + wm * 32;
    int n0 = blockIdx.x * 128 + wn * 32;
    f32x8 a00 = zero8(), a01 = zero8(), a10 = zero8(), a11 = zero8();
    for (int k = 0; k < K; k += 32) {
        __builtin_prefetch(A + (size_t)m0 * K + k + 128, 0, 1);  // global_prefetch_b8
        half16 fa0 = load_frag_a(A + (size_t)m0 * K + k, K);
        half16 fa1 = load_frag_a(A + (size_t)(m0 + 16) * K + k, K);
        half16 fb0 = load_frag_b(W + (size_t)n0 * K + k, K);
        half16 fb1 = load_frag_b(W + (size_t)(n0 + 16) * K + k, K);
        a00 = wmma_f16(fa0, fb0, a00);
        a01 = wmma_f16(fa0, fb1, a01);
        a10 = wmma_f16(fa1, fb0, a10);
        a11 = wmma_f16(fa1, fb1, a11);
    }
    store_frag(C, ldc, m0,      n0,      a00, Nstore, residual);
    store_frag(C, ldc, m0,      n0 + 16, a01, Nstore, residual);
    store_frag(C, ldc, m0 + 16, n0,      a10, Nstore, residual);
    store_frag(C, ldc, m0 + 16, n0 + 16, a11, Nstore, residual);
}

// ---------------- GEGLU split: h -> qh/kh (K=64 pad), loc -> cat[:, :384], val -> valT ----------------
__global__ __launch_bounds__(256) void geglu_kernel(const float* __restrict__ h,
                                                    _Float16* __restrict__ qh,
                                                    _Float16* __restrict__ kh,
                                                    _Float16* __restrict__ valT,
                                                    _Float16* __restrict__ cat) {
    int row = blockIdx.x;
    int tid = threadIdx.x;
    int b = row / SS, s = row % SS;
    const float* hr = h + (size_t)row * NEXP_P;
    if (tid < QKP) {
        qh[(size_t)row * QKP + tid] = (tid < QKD) ? (_Float16)hr[tid]        : (_Float16)0.0f;
        kh[(size_t)row * QKP + tid] = (tid < QKD) ? (_Float16)hr[QKD + tid]  : (_Float16)0.0f;
    }
    for (int j = tid; j < EE; j += 256) {
        float lin = hr[2 * QKD + j];
        float pg  = hr[2 * QKD + EE + j];
        float gel = 0.5f * pg * (1.0f + erff(pg * 0.70710678118654752f));
        float gg  = lin * gel;
        if (j < EE - DD) {
            cat[(size_t)row * EE + j] = (_Float16)gg;                        // loc
        } else {
            valT[((size_t)b * DD + (j - (EE - DD))) * SS + s] = (_Float16)gg; // val transposed
        }
    }
}

// ---------------- causal flash attention ----------------
// grid = (S/16 query tiles, B). block = 256 = 8 waves. All waves share the same
// 16-query tile; wave w owns value dims [w*48, w*48+48) (3 N-tiles of 16).
// Scores (QK^T) computed redundantly per wave via WMMA; P goes C-layout ->
// LDS -> A-layout for the PV WMMA.
__global__ __launch_bounds__(256) void attn_kernel(const _Float16* __restrict__ qh,
                                                   const _Float16* __restrict__ kh,
                                                   const _Float16* __restrict__ valT,
                                                   _Float16* __restrict__ cat) {
    __shared__ _Float16 pbuf[8 * 16 * 32];
    int wave = threadIdx.x >> 5, lane = threadIdx.x & 31;
    int hf = lane >> 4, col = lane & 15;
    int q0 = blockIdx.x * 16;
    int b  = blockIdx.y;
    const _Float16* qh_b = qh   + (size_t)b * SS * QKP;
    const _Float16* kh_b = kh   + (size_t)b * SS * QKP;
    const _Float16* vT_b = valT + (size_t)b * DD * SS;
    _Float16*       ct_b = cat  + (size_t)b * SS * EE;

    half16 aq0 = load_frag_a(qh_b + (size_t)q0 * QKP,      QKP);  // K 0..31
    half16 aq1 = load_frag_a(qh_b + (size_t)q0 * QKP + 32, QKP);  // K 32..63 (zero pad)

    float m[8], l[8];
    f32x8 acc[3];
    for (int r = 0; r < 8; ++r) { m[r] = -3e30f; l[r] = 0.0f; }
    for (int t = 0; t < 3; ++t) acc[t] = zero8();

    const float isq = 0.14433756729740643f;  // 1/sqrt(48)
    int ntiles = (q0 + 16 + 31) >> 5;
    _Float16* pw = pbuf + wave * 512;

    for (int kt = 0; kt < ntiles; ++kt) {
        int k0 = kt * 32;
        f32x8 s0 = zero8(), s1 = zero8();
        half16 b00 = load_frag_b(kh_b + (size_t)k0 * QKP,             QKP);
        half16 b01 = load_frag_b(kh_b + (size_t)k0 * QKP + 32,        QKP);
        half16 b10 = load_frag_b(kh_b + (size_t)(k0 + 16) * QKP,      QKP);
        half16 b11 = load_frag_b(kh_b + (size_t)(k0 + 16) * QKP + 32, QKP);
        s0 = wmma_f16(aq0, b00, s0); s0 = wmma_f16(aq1, b01, s0);
        s1 = wmma_f16(aq0, b10, s1); s1 = wmma_f16(aq1, b11, s1);

        float p0[8], p1[8], sc[8];
        for (int r = 0; r < 8; ++r) {
            int qrow = q0 + r + hf * 8;
            float v0 = (k0 + col      <= qrow) ? s0[r] * isq : -3e30f;
            float v1 = (k0 + 16 + col <= qrow) ? s1[r] * isq : -3e30f;
            float mt = fmaxf(v0, v1);
            for (int o = 1; o < 16; o <<= 1) mt = fmaxf(mt, __shfl_xor(mt, o, 32));
            float mn = fmaxf(m[r], mt);
            sc[r] = __expf(m[r] - mn);
            m[r] = mn;
            p0[r] = __expf(v0 - mn);
            p1[r] = __expf(v1 - mn);
            float rsum = p0[r] + p1[r];
            for (int o = 1; o < 16; o <<= 1) rsum += __shfl_xor(rsum, o, 32);
            l[r] = l[r] * sc[r] + rsum;
        }
        // C-layout -> row-major 16x32 f16 tile in this wave's LDS slice
        for (int r = 0; r < 8; ++r) {
            int row = r + hf * 8;
            pw[row * 32 + col]      = (_Float16)p0[r];
            pw[row * 32 + 16 + col] = (_Float16)p1[r];
        }
        __syncthreads();   // uniform: ntiles identical for every wave in block
        half16 pa = load_frag_a(pw, 32);
        for (int t = 0; t < 3; ++t) {
            int n0 = wave * 48 + t * 16;
            half16 bv = load_frag_b(vT_b + (size_t)n0 * SS + k0, SS);
            for (int r = 0; r < 8; ++r) acc[t][r] *= sc[r];
            acc[t] = wmma_f16(pa, bv, acc[t]);
        }
        __syncthreads();
    }
    for (int t = 0; t < 3; ++t)
        for (int r = 0; r < 8; ++r) {
            int row = q0 + r + hf * 8;
            int c   = (EE - DD) + wave * 48 + t * 16 + col;
            ct_b[(size_t)row * EE + c] = (_Float16)(acc[t][r] / l[r]);
        }
}

// ---------------- host launcher ----------------
extern "C" void kernel_launch(void* const* d_in, const int* in_sizes, int n_in,
                              void* d_out, int out_size, void* d_ws, size_t ws_size,
                              hipStream_t stream) {
    const int*   q      = (const int*)d_in[0];
    const float* freqs  = (const float*)d_in[1];
    const float* exp_w  = (const float*)d_in[2];
    const float* proj_w = (const float*)d_in[3];
    const float* blk_g  = (const float*)d_in[4];
    const float* blk_b  = (const float*)d_in[5];
    const float* ln_g   = (const float*)d_in[6];
    const float* ln_b   = (const float*)d_in[7];
    const float* out_w  = (const float*)d_in[8];
    float* out = (float*)d_out;

    // workspace carve-up (~110 MB total)
    char* wsb = (char*)d_ws;
    size_t off = 0;
    auto alloc = [&](size_t bytes) {
        void* p = wsb + off;
        off = (off + bytes + 255) & ~(size_t)255;
        return p;
    };
    float*     x_f32   = (float*)    alloc((size_t)MM * DD * 4);
    _Float16*  xn_f16  = (_Float16*) alloc((size_t)MM * DD * 2);
    float*     h_f32   = (float*)    alloc((size_t)MM * NEXP_P * 4);
    _Float16*  qh_f16  = (_Float16*) alloc((size_t)MM * QKP * 2);
    _Float16*  kh_f16  = (_Float16*) alloc((size_t)MM * QKP * 2);
    _Float16*  vT_f16  = (_Float16*) alloc((size_t)MM * DD * 2);
    _Float16*  cat_f16 = (_Float16*) alloc((size_t)MM * EE * 2);
    _Float16*  expw16  = (_Float16*) alloc((size_t)8 * NEXP_P * DD * 2);
    _Float16*  projw16 = (_Float16*) alloc((size_t)8 * DD * EE * 2);
    _Float16*  outw16  = (_Float16*) alloc((size_t)NVOC_P * DD * 2);
    (void)ws_size; (void)in_sizes; (void)n_in; (void)out_size;

    // weight conversion / padding
    {
        size_t t1 = (size_t)8 * NEXP_P * DD;
        conv_expw_kernel<<<dim3((t1 + 255) / 256), 256, 0, stream>>>(exp_w, expw16);
        size_t t2 = (size_t)8 * DD * EE;
        conv_projw_kernel<<<dim3((t2 + 255) / 256), 256, 0, stream>>>(proj_w, projw16);
        size_t t3 = (size_t)NVOC_P * DD;
        conv_outw_kernel<<<dim3((t3 + 255) / 256), 256, 0, stream>>>(out_w, outw16);
    }

    // embedding + LN
    embed_ln_kernel<<<dim3(MM), 128, 0, stream>>>(q, freqs, ln_g, ln_b, x_f32);

    for (int ib = 0; ib < 8; ++ib) {
        ln_f16_kernel<<<dim3(MM), 128, 0, stream>>>(x_f32, blk_g + ib * DD, blk_b + ib * DD, xn_f16);
        // h = xn @ ew^T   (M=8192, N=1664 padded, K=384)
        gemm_kernel<<<dim3(NEXP_P / 128, MM / 64), 256, 0, stream>>>(
            xn_f16, expw16 + (size_t)ib * NEXP_P * DD, h_f32, DD, NEXP_P, NEXP_P, 0);
        geglu_kernel<<<dim3(MM), 256, 0, stream>>>(h_f32, qh_f16, kh_f16, vT_f16, cat_f16);
        attn_kernel<<<dim3(SS / 16, BB), 256, 0, stream>>>(qh_f16, kh_f16, vT_f16, cat_f16);
        // x += cat @ pw^T  (M=8192, N=384, K=768) with residual add
        gemm_kernel<<<dim3(DD / 128, MM / 64), 256, 0, stream>>>(
            cat_f16, projw16 + (size_t)ib * DD * EE, x_f32, EE, DD, DD, 1);
    }

    // final LN + logits (N padded 128, store 100)
    ln_f16_kernel<<<dim3(MM), 128, 0, stream>>>(x_f32, ln_g, ln_b, xn_f16);
    gemm_kernel<<<dim3(NVOC_P / 128, MM / 64), 256, 0, stream>>>(
        xn_f16, outw16, out, DD, NVOC, NVOC, 0);
}
